// DeltaNetBlock_37297495998680
// MI455X (gfx1250) — compile-verified, matrix operand order checked by reading, not run on hardware
//
#include <hip/hip_runtime.h>
#include <hip/hip_bf16.h>
#include <math.h>

// ---------------- problem dims ----------------
constexpr int Bc = 4, Tc = 4096, DIMc = 1024, Hc = 12, DKc = 64, DVc = 128;
constexpr int KEYc = Hc * DKc;       // 768
constexpr int VALc = Hc * DVc;       // 1536
constexpr int FFNc = 2816;
constexpr int Mrows = Bc * Tc;       // 16384

typedef __bf16 bf16;
typedef __attribute__((ext_vector_type(16))) __bf16 v16bf;
typedef __attribute__((ext_vector_type(8)))  float   v8f;

// =====================================================================
// Weight convert + transpose: f32 W[K,N] -> bf16 Wt[N,K]
// =====================================================================
__global__ void wtrans_bf16_kernel(const float* __restrict__ w, bf16* __restrict__ wt,
                                   int K, int N) {
  size_t i = (size_t)blockIdx.x * 256 + threadIdx.x;
  if (i >= (size_t)K * N) return;
  int k = (int)(i / N), n = (int)(i % N);
  wt[(size_t)n * K + k] = (bf16)w[i];
}

// =====================================================================
// RMSNorm row (f32 in, bf16 out). One block per row.
// =====================================================================
__global__ __launch_bounds__(256) void rmsnorm_bf16_kernel(
    const float* __restrict__ x, const float* __restrict__ w,
    bf16* __restrict__ out, int ncols) {
  int row = blockIdx.x;
  const float* xr = x + (size_t)row * ncols;
  float ss = 0.f;
  for (int c = threadIdx.x; c < ncols; c += 256) { float v = xr[c]; ss += v * v; }
  __shared__ float sred[8];
  __shared__ float stot;
  for (int o = 16; o > 0; o >>= 1) ss += __shfl_down(ss, o, 32);
  if ((threadIdx.x & 31) == 0) sred[threadIdx.x >> 5] = ss;
  __syncthreads();
  if (threadIdx.x == 0) {
    float t = 0.f;
    for (int i = 0; i < 8; ++i) t += sred[i];
    stot = rsqrtf(t / (float)ncols + 1e-5f);
  }
  __syncthreads();
  float rs = stot;
  for (int c = threadIdx.x; c < ncols; c += 256)
    out[(size_t)row * ncols + c] = (bf16)(xr[c] * rs * w[c]);
}

// =====================================================================
// bf16 WMMA GEMM: C[M,N] = A[M,K] * Bt[N,K]^T   (f32 accumulate)
// 128x128 tile, K-step 64 (2 WMMA k-subs), 8 waves: wave = 2x4 grid of
// 16x16 WMMA tiles. Global->LDS staging uses CDNA5 async-to-LDS copies
// (ASYNCcnt) so data never bounces through VGPRs.
// =====================================================================
__device__ __forceinline__ void gemm_load_tile_async(
    const bf16* __restrict__ g, int row0, int K, int k0, bf16 (*sm)[72], int tid) {
#pragma unroll
  for (int i = 0; i < 4; ++i) {
    int c  = tid + i * 256;        // 0..1023 ; 128 rows * 8 chunks of 8 halfs
    int r  = c >> 3;
    int ch = (c & 7) * 8;
    const bf16* gp = g + (size_t)(row0 + r) * K + k0 + ch;
    unsigned lds = (unsigned)(size_t)&sm[r][ch];   // low 32 bits of flat = LDS byte addr
    asm volatile("global_load_async_to_lds_b128 %0, %1, off"
                 :: "v"(lds), "v"(gp) : "memory");
  }
}

__device__ __forceinline__ void wait_async_all() {
  asm volatile("s_wait_asynccnt 0x0" ::: "memory");
}

__global__ __launch_bounds__(256) void gemm_bf16_kernel(
    const bf16* __restrict__ A, const bf16* __restrict__ Bt,
    float* __restrict__ C, int M, int N, int K) {
  __shared__ bf16 sa[2][128][72];
  __shared__ bf16 sb[2][128][72];
  const int tid  = threadIdx.x;
  const int m0   = blockIdx.y * 128;
  const int n0   = blockIdx.x * 128;
  const int lane = tid & 31, wave = tid >> 5;
  const int wm   = (wave & 3) * 32;   // 4 wave-rows of 32
  const int wn   = (wave >> 2) * 64;  // 2 wave-cols of 64

  v8f acc[2][4];
#pragma unroll
  for (int i = 0; i < 2; ++i)
#pragma unroll
    for (int j = 0; j < 4; ++j)
#pragma unroll
      for (int r = 0; r < 8; ++r) acc[i][j][r] = 0.f;

  const int nk = K >> 6;
  gemm_load_tile_async(A,  m0, K, 0, sa[0], tid);
  gemm_load_tile_async(Bt, n0, K, 0, sb[0], tid);
  wait_async_all();
  __syncthreads();

  const int lm = lane & 15;
  const int kb = (lane >> 4) << 3;  // 0 or 8 (lane-half K base per ISA layout)

  union Frag { v16bf v; uint4 u[2]; };

  for (int kt = 0; kt < nk; ++kt) {
    const int cur = kt & 1;
    if (kt + 1 < nk) {
      gemm_load_tile_async(A,  m0, K, (kt + 1) << 6, sa[cur ^ 1], tid);
      gemm_load_tile_async(Bt, n0, K, (kt + 1) << 6, sb[cur ^ 1], tid);
    }
#pragma unroll
    for (int s = 0; s < 2; ++s) {
      const int ko = s * 32;
      Frag fa[2], fb[4];
#pragma unroll
      for (int i = 0; i < 2; ++i) {
        int r = wm + i * 16 + lm;
        fa[i].u[0] = *(const uint4*)&sa[cur][r][ko + kb];       // K = kb..kb+7
        fa[i].u[1] = *(const uint4*)&sa[cur][r][ko + kb + 16];  // K = kb+16..kb+23
      }
#pragma unroll
      for (int j = 0; j < 4; ++j) {
        int r = wn + j * 16 + lm;
        fb[j].u[0] = *(const uint4*)&sb[cur][r][ko + kb];
        fb[j].u[1] = *(const uint4*)&sb[cur][r][ko + kb + 16];
      }
#pragma unroll
      for (int i = 0; i < 2; ++i)
#pragma unroll
        for (int j = 0; j < 4; ++j)
          acc[i][j] = __builtin_amdgcn_wmma_f32_16x16x32_bf16(
              false, fa[i].v, false, fb[j].v, (short)0, acc[i][j], false, false);
    }
    wait_async_all();
    __syncthreads();
  }

  const int rg = (lane >> 4) << 3;  // C/D layout: lanes 16..31 hold M+8
#pragma unroll
  for (int i = 0; i < 2; ++i)
#pragma unroll
    for (int j = 0; j < 4; ++j) {
      int n  = n0 + wn + j * 16 + lm;
      int mb = m0 + wm + i * 16 + rg;
#pragma unroll
      for (int r = 0; r < 8; ++r)
        C[(size_t)(mb + r) * N + n] = acc[i][j][r];
    }
}

// =====================================================================
// Causal depthwise conv (K=4) + SiLU over [B,T,C] flat
// =====================================================================
__global__ void conv_silu_kernel(const float* __restrict__ xr, const float* __restrict__ cw,
                                 float* __restrict__ y, int C, size_t n) {
  size_t i = (size_t)blockIdx.x * 256 + threadIdx.x;
  if (i >= n) return;
  int c = (int)(i % C);
  int t = (int)((i / C) % Tc);
  float s = 0.f;
#pragma unroll
  for (int j = 0; j < 4; ++j) {
    int tt = t - 3 + j;
    if (tt >= 0) s += xr[i - (size_t)(3 - j) * C] * cw[c * 4 + j];
  }
  y[i] = s / (1.f + __expf(-s));
}

// =====================================================================
// l2norm over rows of 64 (one wave per row), in place, with extra scale
// =====================================================================
__global__ __launch_bounds__(256) void l2norm64_kernel(float* __restrict__ q,
                                                       float scale, int nrows) {
  int row = blockIdx.x * 8 + (threadIdx.x >> 5);
  if (row >= nrows) return;
  int lane = threadIdx.x & 31;
  float* r = q + (size_t)row * 64;
  float a = r[lane], b = r[lane + 32];
  float ss = a * a + b * b;
  for (int o = 16; o > 0; o >>= 1) ss += __shfl_xor(ss, o, 32);
  float rn = rsqrtf(ss + 1e-6f) * scale;
  r[lane] = a * rn;
  r[lane + 32] = b * rn;
}

// =====================================================================
// beta = sigmoid(xn@wb), alpha = exp(-exp(A_log)*softplus(xn@wa+dt_bias))
// one block per token; one wave per head (looped)
// =====================================================================
__global__ __launch_bounds__(256) void beta_alpha_kernel(
    const bf16* __restrict__ xn, const float* __restrict__ wb,
    const float* __restrict__ wa, const float* __restrict__ A_log,
    const float* __restrict__ dt_bias, float* __restrict__ beta,
    float* __restrict__ alpha) {
  int row = blockIdx.x;
  __shared__ float xs[DIMc];
  for (int c = threadIdx.x; c < DIMc; c += 256)
    xs[c] = (float)xn[(size_t)row * DIMc + c];
  __syncthreads();
  int lane = threadIdx.x & 31, wave = threadIdx.x >> 5;
  for (int h = wave; h < Hc; h += 8) {
    float sb = 0.f, sa = 0.f;
    for (int d = lane; d < DIMc; d += 32) {
      float xv = xs[d];
      sb += xv * wb[d * Hc + h];
      sa += xv * wa[d * Hc + h];
    }
    for (int o = 16; o > 0; o >>= 1) {
      sb += __shfl_xor(sb, o, 32);
      sa += __shfl_xor(sa, o, 32);
    }
    if (lane == 0) {
      beta[(size_t)row * Hc + h] = 1.f / (1.f + __expf(-sb));
      float z  = sa + dt_bias[h];
      float sp = (z > 20.f) ? z : log1pf(__expf(z));
      alpha[(size_t)row * Hc + h] = __expf(-__expf(A_log[h]) * sp);
    }
  }
}

// =====================================================================
// Gated delta rule scan. One block per (b,h); S[64][128] f32 in LDS.
// thread = (khalf, vcol): owns 32 rows of one state column.
// =====================================================================
__global__ __launch_bounds__(256) void delta_scan_kernel(
    const float* __restrict__ q, const float* __restrict__ k,
    const float* __restrict__ v, const float* __restrict__ beta,
    const float* __restrict__ alpha, float* __restrict__ o) {
  __shared__ float S[64][128];
  __shared__ float qs[64], ks[64], vs[128], red[128], dred[128];
  const int tid   = threadIdx.x;
  const int vcol  = tid & 127;
  const int khalf = tid >> 7;  // 0 or 1
  const int b = blockIdx.x / Hc, h = blockIdx.x % Hc;
#pragma unroll
  for (int i = 0; i < 32; ++i) S[khalf * 32 + i][vcol] = 0.f;
  __syncthreads();

  for (int t = 0; t < Tc; ++t) {
    const size_t rh = ((size_t)(b * Tc + t)) * Hc + h;
    if (tid < 64)       { qs[tid] = q[rh * 64 + tid]; ks[tid] = k[rh * 64 + tid]; }
    else if (tid < 192) { vs[tid - 64] = v[rh * 128 + (tid - 64)]; }
    __syncthreads();                                   // [A]
    const float at = alpha[rh], bt = beta[rh];
    float p = 0.f;
#pragma unroll
    for (int i = 0; i < 32; ++i) {
      const int kk = khalf * 32 + i;
      float s = S[kk][vcol] * at;                      // S *= alpha
      S[kk][vcol] = s;
      p += ks[kk] * s;                                 // mem partial
    }
    if (khalf) red[vcol] = p;
    __syncthreads();                                   // [B]
    if (!khalf) dred[vcol] = (vs[vcol] - (p + red[vcol])) * bt;  // delta
    __syncthreads();                                   // [C]
    const float delta = dred[vcol];
    float po = 0.f;
#pragma unroll
    for (int i = 0; i < 32; ++i) {
      const int kk = khalf * 32 + i;
      float s = S[kk][vcol] + ks[kk] * delta;          // rank-1 update
      S[kk][vcol] = s;
      po += qs[kk] * s;                                // out partial
    }
    if (khalf) red[vcol] = po;
    __syncthreads();                                   // [D]
    if (!khalf) o[rh * 128 + vcol] = po + red[vcol];
  }
}

// =====================================================================
// og = rmsnorm(o, o_norm_w) * silu(gate)   (bf16 out), wave per (b,t,h)
// =====================================================================
__global__ __launch_bounds__(256) void gated_out_kernel(
    const float* __restrict__ o, const float* __restrict__ gate,
    const float* __restrict__ onw, bf16* __restrict__ og, int nrows) {
  int row = blockIdx.x * 8 + (threadIdx.x >> 5);
  if (row >= nrows) return;
  int lane = threadIdx.x & 31;
  const float* orow = o + (size_t)row * 128;
  const float* grow = gate + (size_t)row * 128;
  float v[4];
  float ss = 0.f;
#pragma unroll
  for (int i = 0; i < 4; ++i) { v[i] = orow[lane + 32 * i]; ss += v[i] * v[i]; }
  for (int off = 16; off > 0; off >>= 1) ss += __shfl_xor(ss, off, 32);
  float rs = rsqrtf(ss / 128.f + 1e-5f);
#pragma unroll
  for (int i = 0; i < 4; ++i) {
    int j = lane + 32 * i;
    float g = grow[j];
    float sil = g / (1.f + __expf(-g));
    og[(size_t)row * 128 + j] = (bf16)(v[i] * rs * onw[j] * sil);
  }
}

// =====================================================================
// elementwise helpers
// =====================================================================
__global__ void add_kernel(const float* __restrict__ a, const float* __restrict__ b,
                           float* __restrict__ out, size_t n) {
  size_t i = (size_t)blockIdx.x * 256 + threadIdx.x;
  if (i < n) out[i] = a[i] + b[i];
}
__global__ void add_inplace_kernel(float* __restrict__ out, const float* __restrict__ b,
                                   size_t n) {
  size_t i = (size_t)blockIdx.x * 256 + threadIdx.x;
  if (i < n) out[i] += b[i];
}
__global__ void silu_mul_kernel(const float* __restrict__ u1, const float* __restrict__ u3,
                                bf16* __restrict__ a, size_t n) {
  size_t i = (size_t)blockIdx.x * 256 + threadIdx.x;
  if (i < n) {
    float x = u1[i];
    a[i] = (bf16)((x / (1.f + __expf(-x))) * u3[i]);
  }
}

// =====================================================================
// host-side launcher
// =====================================================================
static inline dim3 g1(size_t n) { return dim3((unsigned)((n + 255) / 256)); }

extern "C" void kernel_launch(void* const* d_in, const int* in_sizes, int n_in,
                              void* d_out, int out_size, void* d_ws, size_t ws_size,
                              hipStream_t stream) {
  (void)in_sizes; (void)n_in; (void)ws_size;
  const float* x       = (const float*)d_in[0];
  const float* anw     = (const float*)d_in[1];
  const float* wq      = (const float*)d_in[2];
  const float* wk      = (const float*)d_in[3];
  const float* wv      = (const float*)d_in[4];
  const float* wb      = (const float*)d_in[5];
  const float* wa      = (const float*)d_in[6];
  const float* A_log   = (const float*)d_in[7];
  const float* dt_bias = (const float*)d_in[8];
  const float* conv_q  = (const float*)d_in[9];
  const float* conv_k  = (const float*)d_in[10];
  const float* conv_v  = (const float*)d_in[11];
  const float* wg      = (const float*)d_in[12];
  const float* onw     = (const float*)d_in[13];
  const float* wo      = (const float*)d_in[14];
  const float* fnw     = (const float*)d_in[15];
  const float* w1      = (const float*)d_in[16];
  const float* w3      = (const float*)d_in[17];
  const float* w2      = (const float*)d_in[18];
  float* out = (float*)d_out;
  (void)out_size;

  char* ws = (char*)d_ws;
  size_t off = 0;
  auto alloc = [&](size_t bytes) { size_t o = off; off = (off + bytes + 255) & ~(size_t)255; return o; };

  // persistent bf16 transposed weights
  bf16* wq_t = (bf16*)(ws + alloc((size_t)KEYc * DIMc * 2));
  bf16* wk_t = (bf16*)(ws + alloc((size_t)KEYc * DIMc * 2));
  bf16* wv_t = (bf16*)(ws + alloc((size_t)VALc * DIMc * 2));
  bf16* wg_t = (bf16*)(ws + alloc((size_t)VALc * DIMc * 2));
  bf16* wo_t = (bf16*)(ws + alloc((size_t)DIMc * VALc * 2));
  bf16* w1_t = (bf16*)(ws + alloc((size_t)FFNc * DIMc * 2));
  bf16* w3_t = (bf16*)(ws + alloc((size_t)FFNc * DIMc * 2));
  bf16* w2_t = (bf16*)(ws + alloc((size_t)DIMc * FFNc * 2));
  // activations
  bf16*  xn    = (bf16*)(ws + alloc((size_t)Mrows * DIMc * 2));   // later reused as hn
  size_t r1sz  = (size_t)Mrows * (KEYc + KEYc + VALc) * 4;        // 192MB (>= M*FFN*4)
  char*  reg1  = ws + alloc(r1sz);
  char*  reg2  = ws + alloc(r1sz);
  float* hbuf  = (float*)(ws + alloc((size_t)Mrows * DIMc * 4));  // attn-out then h
  bf16*  reg4  = (bf16*)(ws + alloc((size_t)Mrows * FFNc * 2));   // og then a
  float* beta  = (float*)(ws + alloc((size_t)Mrows * Hc * 4));
  float* alpha = (float*)(ws + alloc((size_t)Mrows * Hc * 4));

  // region1/2 views
  float* q_raw = (float*)reg1;
  float* k_raw = (float*)(reg1 + (size_t)Mrows * KEYc * 4);
  float* v_raw = (float*)(reg1 + (size_t)Mrows * KEYc * 8);
  float* qc    = (float*)reg2;
  float* kc    = (float*)(reg2 + (size_t)Mrows * KEYc * 4);
  float* vc    = (float*)(reg2 + (size_t)Mrows * KEYc * 8);
  float* obuf  = (float*)reg1;                                    // after conv, reuse
  float* gate  = (float*)(reg1 + (size_t)Mrows * VALc * 4);
  float* u1    = (float*)reg1;                                    // FFN phase
  float* u3    = (float*)reg2;

  // 1) weight convert/transpose
  wtrans_bf16_kernel<<<g1((size_t)DIMc * KEYc), 256, 0, stream>>>(wq, wq_t, DIMc, KEYc);
  wtrans_bf16_kernel<<<g1((size_t)DIMc * KEYc), 256, 0, stream>>>(wk, wk_t, DIMc, KEYc);
  wtrans_bf16_kernel<<<g1((size_t)DIMc * VALc), 256, 0, stream>>>(wv, wv_t, DIMc, VALc);
  wtrans_bf16_kernel<<<g1((size_t)DIMc * VALc), 256, 0, stream>>>(wg, wg_t, DIMc, VALc);
  wtrans_bf16_kernel<<<g1((size_t)VALc * DIMc), 256, 0, stream>>>(wo, wo_t, VALc, DIMc);
  wtrans_bf16_kernel<<<g1((size_t)DIMc * FFNc), 256, 0, stream>>>(w1, w1_t, DIMc, FFNc);
  wtrans_bf16_kernel<<<g1((size_t)DIMc * FFNc), 256, 0, stream>>>(w3, w3_t, DIMc, FFNc);
  wtrans_bf16_kernel<<<g1((size_t)FFNc * DIMc), 256, 0, stream>>>(w2, w2_t, FFNc, DIMc);

  // 2) xn = rmsnorm(x)
  rmsnorm_bf16_kernel<<<Mrows, 256, 0, stream>>>(x, anw, xn, DIMc);

  // 3) q/k/v projections (WMMA GEMMs)
  gemm_bf16_kernel<<<dim3(KEYc / 128, Mrows / 128), 256, 0, stream>>>(xn, wq_t, q_raw, Mrows, KEYc, DIMc);
  gemm_bf16_kernel<<<dim3(KEYc / 128, Mrows / 128), 256, 0, stream>>>(xn, wk_t, k_raw, Mrows, KEYc, DIMc);
  gemm_bf16_kernel<<<dim3(VALc / 128, Mrows / 128), 256, 0, stream>>>(xn, wv_t, v_raw, Mrows, VALc, DIMc);

  // 4) short conv + silu
  conv_silu_kernel<<<g1((size_t)Mrows * KEYc), 256, 0, stream>>>(q_raw, conv_q, qc, KEYc, (size_t)Mrows * KEYc);
  conv_silu_kernel<<<g1((size_t)Mrows * KEYc), 256, 0, stream>>>(k_raw, conv_k, kc, KEYc, (size_t)Mrows * KEYc);
  conv_silu_kernel<<<g1((size_t)Mrows * VALc), 256, 0, stream>>>(v_raw, conv_v, vc, VALc, (size_t)Mrows * VALc);

  // 5) l2norm q (×DK^-1/2), k
  l2norm64_kernel<<<(Mrows * Hc) / 8, 256, 0, stream>>>(qc, 0.125f, Mrows * Hc);
  l2norm64_kernel<<<(Mrows * Hc) / 8, 256, 0, stream>>>(kc, 1.0f, Mrows * Hc);

  // 6) beta / alpha
  beta_alpha_kernel<<<Mrows, 256, 0, stream>>>(xn, wb, wa, A_log, dt_bias, beta, alpha);

  // 8) gate GEMM first (region1 raw buffers now free -> o/gate live there)
  gemm_bf16_kernel<<<dim3(VALc / 128, Mrows / 128), 256, 0, stream>>>(xn, wg_t, gate, Mrows, VALc, DIMc);

  // 7) gated delta-rule scan
  delta_scan_kernel<<<Bc * Hc, 256, 0, stream>>>(qc, kc, vc, beta, alpha, obuf);

  // 9) og = rmsnorm(o)*silu(gate)  (bf16)
  gated_out_kernel<<<(Mrows * Hc) / 8, 256, 0, stream>>>(obuf, gate, onw, reg4, Mrows * Hc);

  // 10) attn out projection; 11) h = x + att
  gemm_bf16_kernel<<<dim3(DIMc / 128, Mrows / 128), 256, 0, stream>>>(reg4, wo_t, hbuf, Mrows, DIMc, VALc);
  add_kernel<<<g1((size_t)Mrows * DIMc), 256, 0, stream>>>(x, hbuf, hbuf, (size_t)Mrows * DIMc);

  // 12) hn = rmsnorm(h) (reuse xn slot)
  rmsnorm_bf16_kernel<<<Mrows, 256, 0, stream>>>(hbuf, fnw, xn, DIMc);

  // 13) FFN up projections
  gemm_bf16_kernel<<<dim3(FFNc / 128, Mrows / 128), 256, 0, stream>>>(xn, w1_t, u1, Mrows, FFNc, DIMc);
  gemm_bf16_kernel<<<dim3(FFNc / 128, Mrows / 128), 256, 0, stream>>>(xn, w3_t, u3, Mrows, FFNc, DIMc);

  // 14) a = silu(u1)*u3 (bf16, reuse reg4)
  silu_mul_kernel<<<g1((size_t)Mrows * FFNc), 256, 0, stream>>>(u1, u3, reg4, (size_t)Mrows * FFNc);

  // 15) down projection into d_out; 16) += h
  gemm_bf16_kernel<<<dim3(DIMc / 128, Mrows / 128), 256, 0, stream>>>(reg4, w2_t, out, Mrows, DIMc, FFNc);
  add_inplace_kernel<<<g1((size_t)Mrows * DIMc), 256, 0, stream>>>(out, hbuf, (size_t)Mrows * DIMc);
}
